// MultiHeadAttention_69741678952760
// MI455X (gfx1250) — compile-verified, matrix operand order checked by reading, not run on hardware
//
#include <hip/hip_runtime.h>
#include <hip/hip_bf16.h>
#include <math.h>

typedef __attribute__((ext_vector_type(16))) __bf16 v16bf;
typedef __attribute__((ext_vector_type(8)))  float  v8f;
typedef __attribute__((ext_vector_type(4)))  unsigned u32x4;
typedef __attribute__((ext_vector_type(8)))  int      i32x8;
typedef __attribute__((ext_vector_type(4)))  int      i32x4;

#define BATCH  4
#define SEQ    2048
#define DMODEL 512
#define NHEADS 8
#define DHEAD  64
#define MTOT   (BATCH * SEQ)   // 8192 total rows
#define LOG2E  1.4426950408889634f

static __device__ __forceinline__ v8f zero8() {
  v8f z = {0.f, 0.f, 0.f, 0.f, 0.f, 0.f, 0.f, 0.f};
  return z;
}

static __device__ __forceinline__ v8f wmma_bf16(v16bf a, v16bf b, v8f c) {
  // (neg_a, A, neg_b, B, c_mod, C, reuse_a, reuse_b)
  return __builtin_amdgcn_wmma_f32_16x16x32_bf16(false, a, false, b, (short)0, c, false, false);
}

// xor-butterfly within 16-lane groups via ds_swizzle immediate pattern:
// offset = {xor_mask[14:10], or_mask[9:5]=0, and_mask[4:0]=0x1f}
#define SWZ_XOR(v, M) \
  __int_as_float(__builtin_amdgcn_ds_swizzle(__float_as_int(v), (((M) << 10) | 0x1f)))

// Async global -> LDS copy of 16 bytes per lane (gfx1250 path, ASYNCcnt).
static __device__ __forceinline__ void async_load_b128(void* lds_dst, const void* gsrc) {
  const unsigned lds_off = (unsigned)(uintptr_t)lds_dst;           // low 32b of flat = DS addr
  const unsigned long long ga = (unsigned long long)(uintptr_t)gsrc;
  asm volatile("global_load_async_to_lds_b128 %0, %1, off"
               :: "v"(lds_off), "v"(ga) : "memory");
}
static __device__ __forceinline__ void wait_async0() {
  asm volatile("s_wait_asynccnt 0" ::: "memory");
}

// Tensor Data Mover: DMA a 2D bf16 tile [tile_d1 rows x tile_d0 elems] from a
// row-major tensor (row stride = stride0 elems) into LDS, packed contiguously.
// D# built per CDNA5 ISA ch.8: group0 = {flags, lds_addr, global_addr, type=2},
// group1 = {mask/data_size, dims, tile dims, strides}; groups 2/3 zero (2D).
static __device__ __forceinline__ void tdm_load_2d(void* lds_dst, const void* gsrc,
                                                   unsigned tensor_d0, unsigned tensor_d1,
                                                   unsigned stride0,
                                                   unsigned tile_d0, unsigned tile_d1) {
  const unsigned long long ga = (unsigned long long)(uintptr_t)gsrc;
  u32x4 g0;
  g0[0] = 1u;                                             // count=1, user D#, no gather
  g0[1] = (unsigned)(uintptr_t)lds_dst;                   // lds_addr (bytes)
  g0[2] = (unsigned)(ga & 0xffffffffu);                   // global_addr[31:0]
  g0[3] = (unsigned)((ga >> 32) & 0x01ffffffu) | (2u << 30); // global_addr[56:32] | type=2
  i32x8 g1;
  g1[0] = (int)(1u << 16);                                // data_size=1 (2B); wg_mask=0; no flags
  g1[1] = (int)((tensor_d0 & 0xffffu) << 16);             // barrier_addr=0 | tensor_dim0[15:0]
  g1[2] = (int)((tensor_d0 >> 16) | ((tensor_d1 & 0xffffu) << 16)); // dim0[31:16] | dim1[15:0]
  g1[3] = (int)((tensor_d1 >> 16) | (tile_d0 << 16));     // dim1[31:16] | tile_dim0
  g1[4] = (int)(tile_d1 & 0xffffu);                       // tile_dim1 | tile_dim2=0
  g1[5] = (int)stride0;                                   // tensor_dim0_stride[31:0]
  g1[6] = 0;                                              // stride0[47:32] | dim1_stride[15:0]
  g1[7] = 0;                                              // dim1_stride[47:16]
  i32x4 z4 = {0, 0, 0, 0};
#if __clang_major__ >= 23
  i32x8 z8 = {0, 0, 0, 0, 0, 0, 0, 0};
  __builtin_amdgcn_tensor_load_to_lds(g0, g1, z4, z4, z8, 0);
#else
  __builtin_amdgcn_tensor_load_to_lds(g0, g1, z4, z4, 0);
#endif
}

// Load a 16x32 bf16 A/B fragment from a row-major [rows, ld] matrix (T = float or __bf16).
// ISA layout: lane<16 -> row=lane,   K = {k0..k0+7, k0+16..k0+23}
//             lane>=16 -> row=lane-16, K = {k0+8..k0+15, k0+24..k0+31}
template<typename T>
static __device__ __forceinline__ v16bf load_frag(const T* __restrict__ base, int row0,
                                                  int ld, int k0, int lane) {
  const T* p = base + (size_t)(row0 + (lane & 15)) * ld + (k0 + ((lane >> 4) << 3));
  v16bf f;
#pragma unroll
  for (int i = 0; i < 8; ++i) f[i] = (__bf16)(float)p[i];
#pragma unroll
  for (int i = 0; i < 8; ++i) f[8 + i] = (__bf16)(float)p[16 + i];
  return f;
}

static __device__ __forceinline__ v16bf load_frag_scaled(const __bf16* __restrict__ base,
                                                         int row0, int ld, int k0,
                                                         int lane, float sc) {
  const __bf16* p = base + (size_t)(row0 + (lane & 15)) * ld + (k0 + ((lane >> 4) << 3));
  v16bf f;
#pragma unroll
  for (int i = 0; i < 8; ++i) f[i] = (__bf16)((float)p[i] * sc);
#pragma unroll
  for (int i = 0; i < 8; ++i) f[8 + i] = (__bf16)((float)p[16 + i] * sc);
  return f;
}

// ---------------------------------------------------------------------------
// Generic 512-K GEMM: out = A[M,512] * W[512(e),512(d)]^T + bias
// STORE_MODE: 0 = head-split bf16 [b,h,l,64], 1 = head-split transposed bf16
//             [b,h,64,l] (for V), 2 = flat f32 [m,512]
// Block = 256 threads = 8 waves; each wave computes a 64x64 tile.
// ---------------------------------------------------------------------------
template<typename AT, int STORE_MODE>
__global__ void __launch_bounds__(256)
gemm512_wmma(const AT* __restrict__ A, const float* __restrict__ W,
             const float* __restrict__ bias, void* __restrict__ outp) {
  const int lane = threadIdx.x & 31;
  const int wv   = threadIdx.x >> 5;
  const int row0 = blockIdx.x * 512 + wv * 64;
  const int col0 = blockIdx.y * 64;

  v8f acc[4][4];
#pragma unroll
  for (int i = 0; i < 4; ++i)
#pragma unroll
    for (int j = 0; j < 4; ++j) acc[i][j] = zero8();

  for (int kk = 0; kk < DMODEL; kk += 32) {
    if (kk + 32 < DMODEL) {   // pull next K-slice toward L2/WGP$ (global_prefetch_b8)
      __builtin_prefetch(A + (size_t)(row0 + (lane & 15)) * DMODEL + kk + 32, 0, 0);
      __builtin_prefetch(W + (size_t)(col0 + (lane & 15)) * DMODEL + kk + 32, 0, 0);
    }
    v16bf a[4], b[4];
#pragma unroll
    for (int i = 0; i < 4; ++i) a[i] = load_frag(A, row0 + 16 * i, DMODEL, kk, lane);
#pragma unroll
    for (int j = 0; j < 4; ++j) b[j] = load_frag(W, col0 + 16 * j, DMODEL, kk, lane);
#pragma unroll
    for (int i = 0; i < 4; ++i)
#pragma unroll
      for (int j = 0; j < 4; ++j) acc[i][j] = wmma_bf16(a[i], b[j], acc[i][j]);
  }

  // C/D layout: VGPR r, lanes 0-15 -> M = r, N = lane; lanes 16-31 -> M = 8+r
  const int n  = lane & 15;
  const int mb = (lane >> 4) << 3;
#pragma unroll
  for (int i = 0; i < 4; ++i) {
#pragma unroll
    for (int j = 0; j < 4; ++j) {
      const int   e  = col0 + 16 * j + n;
      const float bv = bias[e];
#pragma unroll
      for (int r = 0; r < 8; ++r) {
        const int   m = row0 + 16 * i + mb + r;
        const float v = acc[i][j][r] + bv;
        if (STORE_MODE == 2) {
          ((float*)outp)[(size_t)m * DMODEL + e] = v;
        } else {
          const int bb = m >> 11;          // m / SEQ
          const int l  = m & (SEQ - 1);
          const int h  = e >> 6;
          const int d  = e & 63;
          __bf16* o = (__bf16*)outp;
          if (STORE_MODE == 0)
            o[((size_t)(bb * NHEADS + h) * SEQ + l) * DHEAD + d] = (__bf16)v;
          else
            o[((size_t)(bb * NHEADS + h) * DHEAD + d) * SEQ + l] = (__bf16)v;
        }
      }
    }
  }
}

// ---------------------------------------------------------------------------
// Flash attention. Block = 8 waves = 128 query rows of one (b,h).
// K tile (32 keys x 64 dims) staged by the Tensor Data Mover (TENSORcnt),
// V tile (64 dims x 32 keys) staged by async b128 copies (ASYNCcnt);
// both double-buffered and overlapped with WMMA compute.
// Qh/Kh: [b,h,L,64] bf16 ; Vt: [b,h,64,L] bf16 ; att out: [b,L,512] bf16.
// ---------------------------------------------------------------------------
__global__ void __launch_bounds__(256)
attn_wmma(const __bf16* __restrict__ Qh, const __bf16* __restrict__ Kh,
          const __bf16* __restrict__ Vt, const unsigned char* __restrict__ mask,
          __bf16* __restrict__ att) {
  __shared__ __bf16 ktile[2][32 * DHEAD];   // [key][d]   4KB x2  (TDM dest)
  __shared__ __bf16 vtile[2][DHEAD * 32];   // [d][key]   4KB x2  (async dest)
  __shared__ __bf16 pbuf[8 * 16 * 32];      // per-wave P staging, 1KB each

  const int t    = threadIdx.x;
  const int lane = t & 31;
  const int wv   = t >> 5;
  const int b    = blockIdx.z;
  const int h    = blockIdx.y;
  const int m0   = blockIdx.x * 128 + wv * 16;

  const __bf16* Qb = Qh + (size_t)(b * NHEADS + h) * SEQ * DHEAD;
  const __bf16* Kb = Kh + (size_t)(b * NHEADS + h) * SEQ * DHEAD;
  const __bf16* Vb = Vt + (size_t)(b * NHEADS + h) * DHEAD * SEQ;
  const unsigned char* mk = mask + (size_t)b * SEQ;
  __bf16* pw = pbuf + wv * (16 * 32);

  // fold softmax scale AND log2(e) into Q so probabilities are bare exp2
  const float scale = 0.125f * LOG2E;      // 1/sqrt(64) * log2(e)
  const v16bf qa0 = load_frag_scaled(Qb, m0, DHEAD, 0,  lane, scale);
  const v16bf qa1 = load_frag_scaled(Qb, m0, DHEAD, 32, lane, scale);

  float rmax[8], rsum[8];
  v8f o[4];
#pragma unroll
  for (int r = 0; r < 8; ++r) { rmax[r] = -INFINITY; rsum[r] = 0.f; }
#pragma unroll
  for (int j = 0; j < 4; ++j) o[j] = zero8();

  const int n  = lane & 15;
  const int mb = (lane >> 4) << 3;

  // K tile via TDM (one wave issues the DMA descriptor; EXEC is ignored by TDM)
  auto stageK = [&](int buf, int k0) {
    if (wv == 0)
      tdm_load_2d(&ktile[buf][0], Kb + (size_t)k0 * DHEAD,
                  /*tensor_d0=*/DHEAD, /*tensor_d1=*/SEQ, /*stride0=*/DHEAD,
                  /*tile_d0=*/DHEAD, /*tile_d1=*/32);
  };
  // V tile via async copies: 256 threads x 16B; chunk t: d = t>>2, seg = t&3
  auto stageV = [&](int buf, int k0) {
    async_load_b128(&vtile[buf][(t >> 2) * 32 + (t & 3) * 8],
                    Vb + (size_t)(t >> 2) * SEQ + k0 + (t & 3) * 8);
  };

  stageK(0, 0);
  stageV(0, 0);

  const int NT = SEQ / 32;
  for (int kt = 0; kt < NT; ++kt) {
    wait_async0();                                   // own V-tile copies landed
    if (wv == 0) __builtin_amdgcn_s_wait_tensorcnt(0); // K-tile DMA complete
    __syncthreads();                                 // all staging visible; prev tile consumed
    if (kt + 1 < NT) {
      stageK((kt + 1) & 1, (kt + 1) * 32);
      stageV((kt + 1) & 1, (kt + 1) * 32);
    }

    const __bf16* kl = &ktile[kt & 1][0];
    const __bf16* vl = &vtile[kt & 1][0];
    const int k0 = kt * 32;

    // S = (Q*scale*log2e) . K^T for a 16x32 tile (two 16x16 C tiles)
    v8f s[2];
#pragma unroll
    for (int nt = 0; nt < 2; ++nt) {
      v16bf kb0 = load_frag(kl, nt * 16, DHEAD, 0,  lane);
      v16bf kb1 = load_frag(kl, nt * 16, DHEAD, 32, lane);
      v8f tt = zero8();
      tt = wmma_bf16(qa0, kb0, tt);
      tt = wmma_bf16(qa1, kb1, tt);
      const bool padded = mk[k0 + nt * 16 + n] != 0;   // padding mask on key column
#pragma unroll
      for (int r = 0; r < 8; ++r) tt[r] = padded ? -1e9f : tt[r];
      s[nt] = tt;
    }

    // Online softmax (log2 domain). Row r of a C tile lives across 16 lanes.
#pragma unroll
    for (int r = 0; r < 8; ++r) {
      float mx = fmaxf(s[0][r], s[1][r]);
      mx = fmaxf(mx, SWZ_XOR(mx, 1));
      mx = fmaxf(mx, SWZ_XOR(mx, 2));
      mx = fmaxf(mx, SWZ_XOR(mx, 4));
      mx = fmaxf(mx, SWZ_XOR(mx, 8));
      const float nm   = fmaxf(rmax[r], mx);
      const float corr = exp2f(rmax[r] - nm);
      const float p0 = exp2f(s[0][r] - nm);
      const float p1 = exp2f(s[1][r] - nm);
      s[0][r] = p0; s[1][r] = p1;
      float ps = p0 + p1;
      ps += SWZ_XOR(ps, 1);
      ps += SWZ_XOR(ps, 2);
      ps += SWZ_XOR(ps, 4);
      ps += SWZ_XOR(ps, 8);
      rsum[r] = rsum[r] * corr + ps;
      rmax[r] = nm;
#pragma unroll
      for (int j = 0; j < 4; ++j) o[j][r] *= corr;
    }

    // C-layout -> A-layout transpose of P through per-wave LDS buffer
    // (LDS ops of one wave complete in order; fence with s_wait_dscnt)
    asm volatile("s_wait_dscnt 0" ::: "memory");
#pragma unroll
    for (int nt = 0; nt < 2; ++nt)
#pragma unroll
      for (int r = 0; r < 8; ++r)
        pw[(mb + r) * 32 + nt * 16 + n] = (__bf16)s[nt][r];
    asm volatile("s_wait_dscnt 0" ::: "memory");

    v16bf pa;
    {
      const __bf16* pp = pw + (lane & 15) * 32 + ((lane >> 4) << 3);
#pragma unroll
      for (int i = 0; i < 8; ++i) pa[i] = pp[i];
#pragma unroll
      for (int i = 0; i < 8; ++i) pa[8 + i] = pp[16 + i];
    }

    // O += P * V  (V tile in LDS is [d, 32keys] row-major = NxK B operand)
#pragma unroll
    for (int j = 0; j < 4; ++j) {
      v16bf vb = load_frag(vl, j * 16, 32, 0, lane);
      o[j] = wmma_bf16(pa, vb, o[j]);
    }
  }

  // Normalize and write concat-head layout att[b, l, h*64 + d] (bf16)
#pragma unroll
  for (int j = 0; j < 4; ++j) {
#pragma unroll
    for (int r = 0; r < 8; ++r) {
      const int l = m0 + mb + r;
      const int e = h * DHEAD + j * 16 + n;
      att[((size_t)b * SEQ + l) * DMODEL + e] = (__bf16)(o[j][r] / rsum[r]);
    }
  }
}

// ---------------------------------------------------------------------------
// Residual + LayerNorm: out = LN(q + proj) * gamma + beta  (one block per row)
// ---------------------------------------------------------------------------
__global__ void __launch_bounds__(256)
resid_ln(const float* __restrict__ q, const float* __restrict__ proj,
         const float* __restrict__ gamma, const float* __restrict__ beta,
         float* __restrict__ out) {
  __shared__ float red1[256];
  __shared__ float red2[256];
  const int t = threadIdx.x;
  const size_t base = (size_t)blockIdx.x * DMODEL;
  const float x0 = q[base + t]       + proj[base + t];
  const float x1 = q[base + t + 256] + proj[base + t + 256];
  red1[t] = x0 + x1;
  red2[t] = x0 * x0 + x1 * x1;
  __syncthreads();
  for (int sft = 128; sft > 0; sft >>= 1) {
    if (t < sft) { red1[t] += red1[t + sft]; red2[t] += red2[t + sft]; }
    __syncthreads();
  }
  const float mu  = red1[0] * (1.0f / DMODEL);
  const float var = red2[0] * (1.0f / DMODEL) - mu * mu;
  const float inv = rsqrtf(var + 1e-6f);
  out[base + t]       = (x0 - mu) * inv * gamma[t]       + beta[t];
  out[base + t + 256] = (x1 - mu) * inv * gamma[t + 256] + beta[t + 256];
}

extern "C" void kernel_launch(void* const* d_in, const int* in_sizes, int n_in,
                              void* d_out, int out_size, void* d_ws, size_t ws_size,
                              hipStream_t stream) {
  const float* q  = (const float*)d_in[0];
  const float* k  = (const float*)d_in[1];
  const float* v  = (const float*)d_in[2];
  const unsigned char* mask = (const unsigned char*)d_in[3];  // jax bool_ = 1 byte
  const float* Wq = (const float*)d_in[4];
  const float* bq = (const float*)d_in[5];
  const float* Wk = (const float*)d_in[6];
  const float* bk = (const float*)d_in[7];
  const float* Wv = (const float*)d_in[8];
  const float* bv = (const float*)d_in[9];
  const float* Wo = (const float*)d_in[10];
  const float* bo = (const float*)d_in[11];
  const float* gamma = (const float*)d_in[12];
  const float* beta  = (const float*)d_in[13];

  // Workspace carve-up (all bf16 intermediates stay L2-resident):
  char* ws = (char*)d_ws;
  const size_t HEADELEMS = (size_t)BATCH * NHEADS * SEQ * DHEAD;   // 4,194,304
  __bf16* Qh  = (__bf16*)ws;  ws += HEADELEMS * 2;                 // 8 MB
  __bf16* Kh  = (__bf16*)ws;  ws += HEADELEMS * 2;                 // 8 MB
  __bf16* Vt  = (__bf16*)ws;  ws += HEADELEMS * 2;                 // 8 MB
  __bf16* att = (__bf16*)ws;  ws += (size_t)MTOT * DMODEL * 2;     // 8 MB
  float*  prj = (float*)ws;                                        // 16 MB

  const dim3 blk(256);
  const dim3 ggrd(MTOT / 512, DMODEL / 64, 1);     // 16 x 8 blocks
  hipLaunchKernelGGL((gemm512_wmma<float, 0>), ggrd, blk, 0, stream, q, Wq, bq, (void*)Qh);
  hipLaunchKernelGGL((gemm512_wmma<float, 0>), ggrd, blk, 0, stream, k, Wk, bk, (void*)Kh);
  hipLaunchKernelGGL((gemm512_wmma<float, 1>), ggrd, blk, 0, stream, v, Wv, bv, (void*)Vt);

  const dim3 agrd(SEQ / 128, NHEADS, BATCH);       // 16 x 8 x 4
  hipLaunchKernelGGL(attn_wmma, agrd, blk, 0, stream, Qh, Kh, Vt, mask, att);

  hipLaunchKernelGGL((gemm512_wmma<__bf16, 2>), ggrd, blk, 0, stream, att, Wo, bo, (void*)prj);

  hipLaunchKernelGGL(resid_ln, dim3(MTOT), blk, 0, stream, q, prj, gamma, beta, (float*)d_out);
}
